// Int8GPT2Block_6511170421643
// MI455X (gfx1250) — 8-bit matrix operand order repaired and checked by reading, compiled, not run on hardware
//
#include <hip/hip_runtime.h>
#include <hip/hip_bf16.h>
#include <math.h>

#define BB    4
#define SS    1024
#define HH    2048
#define NHH   16
#define HDD   128
#define INNER 8192
#define MTOT  (BB * SS)
#define EPSV  1e-5f

#define S_X   0.02f
#define S_QKV 0.03f
#define S_PV  0.02f
#define S_FC1 0.02f
#define S_FC2 0.03f
#define S_Wq  0.01f

typedef __attribute__((ext_vector_type(8))) int v8i;
typedef __attribute__((address_space(3))) signed char lds_i8;

__device__ __forceinline__ v8i v8zero() {
  v8i z;
#pragma unroll
  for (int i = 0; i < 8; ++i) z[i] = 0;
  return z;
}

__device__ __forceinline__ v8i wmma_i8(v8i a, v8i b, v8i c) {
  return __builtin_amdgcn_wmma_i32_16x16x64_iu8(true, a, true, b, c, false, false);
}

__device__ __forceinline__ signed char q8f(float x) {
  float r = rintf(x);
  r = fminf(127.f, fmaxf(-128.f, r));
  return (signed char)(int)r;
}

__device__ __forceinline__ void async_copy_b128(const signed char* gsrc, signed char* ldst) {
  unsigned loff = (unsigned)(unsigned long long)(lds_i8*)ldst;
  unsigned long long g = (unsigned long long)gsrc;
  asm volatile("global_load_async_to_lds_b128 %0, %1, off" :: "v"(loff), "v"(g) : "memory");
}
__device__ __forceinline__ void wait_asynccnt_le6() { asm volatile("s_wait_asynccnt 0x6" ::: "memory"); }
__device__ __forceinline__ void wait_asynccnt_0()   { asm volatile("s_wait_asynccnt 0x0" ::: "memory"); }

__device__ __forceinline__ v8i load_a_frag(const signed char* base, int lda, int lane) {
  const int l = lane & 15, g = lane >> 4;
  const signed char* rp = base + (size_t)l * lda + 8 * g;
  v8i a;
#pragma unroll
  for (int c = 0; c < 4; ++c) {
    int2 t = *(const int2*)(rp + 16 * c);
    a[2 * c]     = t.x;
    a[2 * c + 1] = t.y;
  }
  return a;
}

__device__ __forceinline__ v8i load_b_frag(const signed char* base, int ldb, int lane) {
  const int n = lane & 15, g = lane >> 4;
  const signed char* rp = base + (size_t)n * ldb + 8 * g;
  v8i b;
#pragma unroll
  for (int c = 0; c < 4; ++c) {
    int2 t = *(const int2*)(rp + 16 * c);
    b[2 * c]     = t.x;
    b[2 * c + 1] = t.y;
  }
  return b;
}

__global__ void quant_w_kernel(const float* __restrict__ w, signed char* __restrict__ o, int n) {
  int i = blockIdx.x * blockDim.x + threadIdx.x;
  int stride = gridDim.x * blockDim.x;
  for (; i < n; i += stride) o[i] = q8f(w[i]);
}

__global__ __launch_bounds__(256) void ln_q_kernel(const float* __restrict__ x,
                                                   const float* __restrict__ gw,
                                                   const float* __restrict__ bw,
                                                   signed char* __restrict__ outq,
                                                   float inv_s) {
  const int row = blockIdx.x;
  const float* xr = x + (size_t)row * HH;
  float s = 0.f, ss = 0.f;
  for (int i = threadIdx.x; i < HH; i += 256) {
    float v = xr[i];
    s += v; ss += v * v;
  }
#pragma unroll
  for (int d = 1; d < 32; d <<= 1) {
    s  += __shfl_xor(s, d, 32);
    ss += __shfl_xor(ss, d, 32);
  }
  __shared__ float sums[8], ssums[8];
  if ((threadIdx.x & 31) == 0) { sums[threadIdx.x >> 5] = s; ssums[threadIdx.x >> 5] = ss; }
  __syncthreads();
  float ts = 0.f, tss = 0.f;
#pragma unroll
  for (int i = 0; i < 8; ++i) { ts += sums[i]; tss += ssums[i]; }
  const float mu   = ts * (1.0f / HH);
  const float var  = tss * (1.0f / HH) - mu * mu;
  const float rstd = rsqrtf(var + EPSV);
  signed char* orow = outq + (size_t)row * HH;
  for (int i = threadIdx.x; i < HH; i += 256) {
    float y = (xr[i] - mu) * rstd * gw[i] + bw[i];
    orow[i] = q8f(y * inv_s);
  }
}

#define TK 64
#define RST 80

template <int EPI>
__global__ __launch_bounds__(128) void gemm_i8_kernel(
    const signed char* __restrict__ A, int lda,
    const signed char* __restrict__ Bw, int ldb,
    const float* __restrict__ bias,
    const float* __restrict__ res,
    float* __restrict__ outf,
    signed char* __restrict__ out8,
    signed char* __restrict__ qo,
    signed char* __restrict__ ko,
    signed char* __restrict__ vo,
    int M, int N, int K, float alpha) {
  __shared__ __align__(16) signed char At[2][128 * RST];
  __shared__ __align__(16) signed char Bt[2][64 * RST];

  const int tid  = threadIdx.x;
  const int lane = tid & 31;
  const int wave = tid >> 5;
  const int wm = wave >> 1, wn = wave & 1;
  const int mblk = blockIdx.y * 128;
  const int nblk = blockIdx.x * 64;

  const signed char* Ab = A + (size_t)mblk * lda;
  const signed char* Bb = Bw + (size_t)nblk * ldb;

  const int r4 = tid >> 2;
  const int c4 = (tid & 3) * 16;

  v8i acc[4][2];
#pragma unroll
  for (int i = 0; i < 4; ++i)
#pragma unroll
    for (int j = 0; j < 2; ++j) acc[i][j] = v8zero();

  const int nk = K / TK;

  {
#pragma unroll
    for (int rr = 0; rr < 4; ++rr) {
      const int row = r4 + 32 * rr;
      async_copy_b128(Ab + (size_t)row * lda + c4, &At[0][row * RST + c4]);
    }
#pragma unroll
    for (int rr = 0; rr < 2; ++rr) {
      const int row = r4 + 32 * rr;
      async_copy_b128(Bb + (size_t)row * ldb + c4, &Bt[0][row * RST + c4]);
    }
  }

  for (int kt = 0; kt < nk; ++kt) {
    const int buf = kt & 1;
    if (kt + 1 < nk) {
      const int k1 = (kt + 1) * TK;
#pragma unroll
      for (int rr = 0; rr < 4; ++rr) {
        const int row = r4 + 32 * rr;
        async_copy_b128(Ab + (size_t)row * lda + k1 + c4, &At[buf ^ 1][row * RST + c4]);
      }
#pragma unroll
      for (int rr = 0; rr < 2; ++rr) {
        const int row = r4 + 32 * rr;
        async_copy_b128(Bb + (size_t)row * ldb + k1 + c4, &Bt[buf ^ 1][row * RST + c4]);
      }
      wait_asynccnt_le6();
    } else {
      wait_asynccnt_0();
    }
    __syncthreads();

    const signed char* Awave = &At[buf][(wm * 64) * RST];
    const signed char* Bwave = &Bt[buf][(wn * 32) * RST];
    v8i af[4];
#pragma unroll
    for (int i = 0; i < 4; ++i)
      af[i] = load_a_frag(Awave + (size_t)(i * 16) * RST, RST, lane);
#pragma unroll
    for (int j = 0; j < 2; ++j) {
      v8i bf = load_b_frag(Bwave + (size_t)(j * 16) * RST, RST, lane);
#pragma unroll
      for (int i = 0; i < 4; ++i)
        acc[i][j] = wmma_i8(af[i], bf, acc[i][j]);
    }
    __syncthreads();
  }

  const int m0 = mblk + wm * 64;
  const int n0 = nblk + wn * 32;
  const int l = lane & 15, g = lane >> 4;
#pragma unroll
  for (int i = 0; i < 4; ++i) {
#pragma unroll
    for (int j = 0; j < 2; ++j) {
#pragma unroll
      for (int r = 0; r < 8; ++r) {
        const int m = m0 + 16 * i + r + 8 * g;
        const int n = n0 + 16 * j + l;
        const float av = (float)acc[i][j][r];
        if (EPI == 0) {
          const signed char val = q8f(av * alpha + bias[n] * alpha);
          const int part = n >> 11;
          const int hn = n & (HH - 1);
          const int h = hn >> 7, d = hn & (HDD - 1);
          const int b = m >> 10, s = m & (SS - 1);
          const size_t bh = (size_t)b * NHH + h;
          if (part == 0)      qo[(bh * SS + s) * HDD + d] = val;
          else if (part == 1) ko[(bh * SS + s) * HDD + d] = val;
          else                vo[(bh * HDD + d) * SS + s] = val;
        } else if (EPI == 1) {
          outf[(size_t)m * N + n] = av * alpha + bias[n] + res[(size_t)m * N + n];
        } else {
          float y = av * alpha + bias[n] * alpha;
          y = fmaxf(y, 0.f);
          out8[(size_t)m * N + n] = q8f(y);
        }
      }
    }
  }
}

__global__ __launch_bounds__(128) void attn_kernel(const signed char* __restrict__ qg,
                                                   const signed char* __restrict__ kg,
                                                   const signed char* __restrict__ vtg,
                                                   signed char* __restrict__ ctx) {
  __shared__ __align__(16) signed char p8s[4][16 * SS];
  const int wave = threadIdx.x >> 5;
  const int lane = threadIdx.x & 31;
  const int tile = blockIdx.x * 4 + wave;
  const int qt = tile & (SS / 16 - 1);
  const int bh = tile / (SS / 16);
  const int q0 = qt * 16;
  signed char* p8 = &p8s[wave][0];

  {
    int4 z; z.x = z.y = z.z = z.w = 0;
    for (int off = lane * 16; off < 16 * SS; off += 32 * 16)
      *(int4*)(p8 + off) = z;
  }
  __syncthreads();

  const signed char* qbase = qg + ((size_t)bh * SS + q0) * HDD;
  const signed char* kbase = kg + (size_t)bh * SS * HDD;
  const signed char* vbase = vtg + (size_t)bh * HDD * SS;
  const int l = lane & 15, g = lane >> 4;

  v8i aq0 = load_a_frag(qbase, HDD, lane);
  v8i aq1 = load_a_frag(qbase + 64, HDD, lane);

  const float dscale = (S_QKV * S_QKV) * (1.0f / 11.313708499f);
  float rm[8], rs[8];
#pragma unroll
  for (int r = 0; r < 8; ++r) { rm[r] = -3.0e38f; rs[r] = 0.f; }

  const int ktmax = q0 >> 4;

  for (int kt = 0; kt <= ktmax; ++kt) {
    v8i sa = wmma_i8(aq0, load_b_frag(kbase + (size_t)kt * 16 * HDD, HDD, lane), v8zero());
    v8i sb = wmma_i8(aq1, load_b_frag(kbase + (size_t)kt * 16 * HDD + 64, HDD, lane), v8zero());
#pragma unroll
    for (int r = 0; r < 8; ++r) {
      const int qi = q0 + r + 8 * g;
      const int kj = kt * 16 + l;
      float sc = (kj <= qi) ? (float)(sa[r] + sb[r]) * dscale : -3.0e38f;
      float tmax = sc;
#pragma unroll
      for (int d = 1; d < 16; d <<= 1) tmax = fmaxf(tmax, __shfl_xor(tmax, d, 32));
      const float nm = fmaxf(rm[r], tmax);
      float e = expf(sc - nm);
#pragma unroll
      for (int d = 1; d < 16; d <<= 1) e += __shfl_xor(e, d, 32);
      rs[r] = rs[r] * expf(rm[r] - nm) + e;
      rm[r] = nm;
    }
  }

  for (int kt = 0; kt <= ktmax; ++kt) {
    v8i sa = wmma_i8(aq0, load_b_frag(kbase + (size_t)kt * 16 * HDD, HDD, lane), v8zero());
    v8i sb = wmma_i8(aq1, load_b_frag(kbase + (size_t)kt * 16 * HDD + 64, HDD, lane), v8zero());
#pragma unroll
    for (int r = 0; r < 8; ++r) {
      const int m = r + 8 * g;
      const int qi = q0 + m;
      const int kj = kt * 16 + l;
      float sc = (kj <= qi) ? (float)(sa[r] + sb[r]) * dscale : -3.0e38f;
      float p = expf(sc - rm[r]) / rs[r];
      int pq = (int)rintf(p * 127.f);
      pq = pq < 0 ? 0 : (pq > 127 ? 127 : pq);
      p8[m * SS + kj] = (signed char)pq;
    }
  }
  __syncthreads();

  const int nchunk = ((q0 + 15) >> 6) + 1;
  v8i cacc[8];
#pragma unroll
  for (int nb = 0; nb < 8; ++nb) cacc[nb] = v8zero();
  for (int c = 0; c < nchunk; ++c) {
    v8i ap = load_a_frag(p8 + c * 64, SS, lane);
#pragma unroll
    for (int nb = 0; nb < 8; ++nb)
      cacc[nb] = wmma_i8(ap, load_b_frag(vbase + (size_t)(nb * 16) * SS + c * 64, SS, lane), cacc[nb]);
  }

  const float apv = (1.0f / 127.0f) * (S_QKV / S_PV);
  const int b = bh / NHH, h = bh % NHH;
#pragma unroll
  for (int nb = 0; nb < 8; ++nb) {
#pragma unroll
    for (int r = 0; r < 8; ++r) {
      const int m = r + 8 * g;
      const int d = nb * 16 + l;
      const int s = q0 + m;
      ctx[((size_t)b * SS + s) * HH + h * HDD + d] = q8f((float)cacc[nb][r] * apv);
    }
  }
}

static inline size_t align256(size_t x) { return (x + 255) & ~(size_t)255; }

extern "C" void kernel_launch(void* const* d_in, const int* in_sizes, int n_in,
                              void* d_out, int out_size, void* d_ws, size_t ws_size,
                              hipStream_t stream) {
  (void)in_sizes; (void)n_in; (void)out_size; (void)ws_size;
  const float* hidden = (const float*)d_in[0];
  const float* g1     = (const float*)d_in[1];
  const float* b1     = (const float*)d_in[2];
  const float* g2     = (const float*)d_in[3];
  const float* b2     = (const float*)d_in[4];
  const float* w_qkv  = (const float*)d_in[5];
  const float* b_qkv  = (const float*)d_in[6];
  const float* w_o    = (const float*)d_in[7];
  const float* b_o    = (const float*)d_in[8];
  const float* w_fc   = (const float*)d_in[9];
  const float* b_fc   = (const float*)d_in[10];
  const float* w_pr   = (const float*)d_in[11];
  const float* b_pr   = (const float*)d_in[12];

  char* ws = (char*)d_ws;
  size_t off = 0;
  signed char* wqkv8 = (signed char*)(ws + off); off = align256(off + (size_t)3 * HH * HH);
  signed char* wo8   = (signed char*)(ws + off); off = align256(off + (size_t)HH * HH);
  signed char* wfc8  = (signed char*)(ws + off); off = align256(off + (size_t)INNER * HH);
  signed char* wpr8  = (signed char*)(ws + off); off = align256(off + (size_t)HH * INNER);
  signed char* xq    = (signed char*)(ws + off); off = align256(off + (size_t)MTOT * HH);
  signed char* q8b   = (signed char*)(ws + off); off = align256(off + (size_t)MTOT * HH);
  signed char* k8b   = (signed char*)(ws + off); off = align256(off + (size_t)MTOT * HH);
  signed char* vt8   = (signed char*)(ws + off); off = align256(off + (size_t)MTOT * HH);
  signed char* ctx8  = (signed char*)(ws + off); off = align256(off + (size_t)MTOT * HH);
  float*       hbuf  = (float*)(ws + off);       off = align256(off + (size_t)MTOT * HH * 4);
  signed char* x2q   = (signed char*)(ws + off); off = align256(off + (size_t)MTOT * HH);
  signed char* fc8   = (signed char*)(ws + off); off = align256(off + (size_t)MTOT * INNER);

  quant_w_kernel<<<2048, 256, 0, stream>>>(w_qkv, wqkv8, 3 * HH * HH);
  quant_w_kernel<<<2048, 256, 0, stream>>>(w_o,   wo8,   HH * HH);
  quant_w_kernel<<<2048, 256, 0, stream>>>(w_fc,  wfc8,  INNER * HH);
  quant_w_kernel<<<2048, 256, 0, stream>>>(w_pr,  wpr8,  HH * INNER);

  ln_q_kernel<<<MTOT, 256, 0, stream>>>(hidden, g1, b1, xq, 1.0f / S_X);

  const float a1 = S_X * S_Wq / S_QKV;
  gemm_i8_kernel<0><<<dim3((3 * HH) / 64, MTOT / 128), 128, 0, stream>>>(
      xq, HH, wqkv8, HH, b_qkv, nullptr, nullptr, nullptr,
      q8b, k8b, vt8, MTOT, 3 * HH, HH, a1);

  attn_kernel<<<(BB * NHH * (SS / 16)) / 4, 128, 0, stream>>>(q8b, k8b, vt8, ctx8);

  gemm_i8_kernel<1><<<dim3(HH / 64, MTOT / 128), 128, 0, stream>>>(
      ctx8, HH, wo8, HH, b_o, hidden, hbuf, nullptr,
      nullptr, nullptr, nullptr, MTOT, HH, HH, S_PV * S_Wq);

  ln_q_kernel<<<MTOT, 256, 0, stream>>>(hbuf, g2, b2, x2q, 1.0f / S_FC1);

  const float afc = S_FC1 * S_Wq / S_FC2;
  gemm_i8_kernel<2><<<dim3(INNER / 64, MTOT / 128), 128, 0, stream>>>(
      x2q, HH, wfc8, HH, b_fc, nullptr, nullptr, fc8,
      nullptr, nullptr, nullptr, MTOT, INNER, HH, afc);

  gemm_i8_kernel<1><<<dim3(HH / 64, MTOT / 128), 128, 0, stream>>>(
      fc8, INNER, wpr8, INNER, b_pr, hbuf, (float*)d_out, nullptr,
      nullptr, nullptr, nullptr, MTOT, HH, INNER, S_FC2 * S_Wq);
}
